// GNNMultiEdgeset_35055523070604
// MI455X (gfx1250) — compile-verified, matrix operand order checked by reading, not run on hardware
//
#include <hip/hip_runtime.h>
#include <math.h>

// Problem constants (from reference): R=1, C=4, N=50000, E=200000, H=128,
// F=128, ED=16, G=64, L=2
#define R_ 1
#define C_ 4
#define N_ 50000
#define E_ 200000
#define H_ 128
#define F_ 128
#define ED_ 16
#define G_ 64
#define L_ 2
#define M_ (R_ * C_ * N_)   // 200000 total node rows

typedef __attribute__((ext_vector_type(16))) _Float16 v16h;
typedef __attribute__((ext_vector_type(8)))  float    v8f;

__device__ __forceinline__ float gelu_exact(float v) {
    // torch F.gelu default (erf form)
    return 0.5f * v * (1.0f + erff(v * 0.7071067811865475f));
}

__device__ __forceinline__ void atomic_add_f32(float* p, float v) {
    // lowers to global_atomic_add_f32 on gfx1250
    unsafeAtomicAdd(p, v);
}

// ---------------------------------------------------------------------------
// Pre-shuffle a 128x128 fp32 weight (row-major, K x N) into the WMMA B-matrix
// lane layout for v_wmma_f32_16x16x32_f16, stored as:
//   frag[((kstep*8 + ntile)*32 + lane)*16 + j]  (f16)
// lane<16 holds K offsets {0..7,16..23}, lane>=16 holds {8..15,24..31} within
// each 32-wide k-step; column n = ntile*16 + (lane&15).
// ---------------------------------------------------------------------------
__global__ void prep_wfrag_kernel(const float* __restrict__ W,
                                  _Float16* __restrict__ dst) {
    int t = blockIdx.x * blockDim.x + threadIdx.x;   // 0..16383
    if (t >= 4 * 8 * 32 * 16) return;
    int j     = t & 15;
    int lane  = (t >> 4) & 31;
    int nt    = (t >> 9) & 7;
    int kstep = t >> 12;
    int hi = lane >> 4;
    int nl = lane & 15;
    int n = nt * 16 + nl;
    int k = kstep * 32 + ((j < 8) ? (hi * 8 + j) : (16 + hi * 8 + (j - 8)));
    dst[t] = (_Float16)W[k * H_ + n];
}

// ---------------------------------------------------------------------------
// Bond encoder: ee[e, :] = edge_attr[e, :16] @ W_be + b_be   (fp32, tiny K)
// One thread per (edge, col); edge_attr load is wave-uniform, W_be L2-resident.
// ---------------------------------------------------------------------------
__global__ void bond_encoder_kernel(const float* __restrict__ ea,
                                    const float* __restrict__ Wbe,
                                    const float* __restrict__ bbe,
                                    float* __restrict__ ee) {
    long g = (long)blockIdx.x * 256 + threadIdx.x;   // over E*H
    if (g >= (long)E_ * H_) return;
    int e   = (int)(g >> 7);
    int col = (int)(g & 127);
    float acc = bbe[col];
#pragma unroll
    for (int k = 0; k < ED_; ++k)
        acc = fmaf(ea[e * ED_ + k], Wbe[k * H_ + col], acc);
    ee[g] = acc;
}

// ---------------------------------------------------------------------------
// Message pass: for each (c, e):
//   agg[c, dst[e], :] += gelu(x[c, src[e], :] + ee[e, :]) * edge_mask[c, e]
// One wave per (c,e); each lane handles a float4 (32 lanes * 4 = 128 = H).
// ---------------------------------------------------------------------------
__global__ void scatter_msg_kernel(const float* __restrict__ xin,
                                   const float* __restrict__ ee,
                                   const int*   __restrict__ ei,
                                   const float* __restrict__ emask,
                                   float* __restrict__ agg) {
    int wid  = threadIdx.x >> 5;
    int lane = threadIdx.x & 31;
    long id = (long)blockIdx.x * 8 + wid;
    if (id >= (long)C_ * E_) return;
    int c = (int)(id / E_);
    int e = (int)(id - (long)c * E_);
    int src = ei[e];
    int dst = ei[E_ + e];
    float em = emask[(long)c * E_ + e];
    const float4 xs = *(const float4*)(xin + ((long)(c * N_ + src)) * H_ + lane * 4);
    const float4 es = *(const float4*)(ee + (long)e * H_ + lane * 4);
    float* ap = agg + ((long)(c * N_ + dst)) * H_ + lane * 4;
    atomic_add_f32(ap + 0, gelu_exact(xs.x + es.x) * em);
    atomic_add_f32(ap + 1, gelu_exact(xs.y + es.y) * em);
    atomic_add_f32(ap + 2, gelu_exact(xs.z + es.z) * em);
    atomic_add_f32(ap + 3, gelu_exact(xs.w + es.w) * em);
}

// ---------------------------------------------------------------------------
// Residual: B = (1 + eps[l]) * xin + B   (float4 per thread)
// ---------------------------------------------------------------------------
__global__ void residual_kernel(const float* __restrict__ xin,
                                const float* __restrict__ eps, int l,
                                float* __restrict__ B) {
    long i = (long)blockIdx.x * 256 + threadIdx.x;
    if (i >= (long)M_ * H_ / 4) return;
    float s = 1.0f + eps[l];
    float4 xv = ((const float4*)xin)[i];
    float4 bv = ((float4*)B)[i];
    bv.x = fmaf(s, xv.x, bv.x);
    bv.y = fmaf(s, xv.y, bv.y);
    bv.z = fmaf(s, xv.z, bv.z);
    bv.w = fmaf(s, xv.w, bv.w);
    ((float4*)B)[i] = bv;
}

// ---------------------------------------------------------------------------
// WMMA GEMM: out[M, 128] = act(in[M, 128] @ W[128,128] + bias)
// 256-thread block = 8 waves; each wave: 16 rows x 128 cols, K=128 as
// 4 k-steps of v_wmma_f32_16x16x32_f16, A converted fp32->f16 in registers,
// B from pre-shuffled fragment buffer (contiguous per-lane 32B loads).
// ---------------------------------------------------------------------------
__global__ __launch_bounds__(256)
void gemm128_kernel(const float* __restrict__ in,
                    const _Float16* __restrict__ wfrag,
                    const float* __restrict__ bias,
                    float* __restrict__ out, int Mrows, int apply_gelu) {
    const int lane = threadIdx.x & 31;
    const int wid  = threadIdx.x >> 5;
    const int lo = lane & 15;
    const int hi = lane >> 4;
    const int rowBase = blockIdx.x * 128 + wid * 16;

    int rA = rowBase + lo;
    if (rA >= Mrows) rA = Mrows - 1;        // clamp tail; stores are guarded

    const v16h* wf = (const v16h*)wfrag;
    v8f acc[8] = {};

#pragma unroll
    for (int ks = 0; ks < 4; ++ks) {
        const int kb = ks * 32;
        const float* pa = in + (long)rA * H_ + kb + hi * 8;
        float4 f0 = ((const float4*)pa)[0];
        float4 f1 = ((const float4*)pa)[1];
        float4 f2 = ((const float4*)(pa + 16))[0];
        float4 f3 = ((const float4*)(pa + 16))[1];
        v16h a;
        a[0]  = (_Float16)f0.x; a[1]  = (_Float16)f0.y;
        a[2]  = (_Float16)f0.z; a[3]  = (_Float16)f0.w;
        a[4]  = (_Float16)f1.x; a[5]  = (_Float16)f1.y;
        a[6]  = (_Float16)f1.z; a[7]  = (_Float16)f1.w;
        a[8]  = (_Float16)f2.x; a[9]  = (_Float16)f2.y;
        a[10] = (_Float16)f2.z; a[11] = (_Float16)f2.w;
        a[12] = (_Float16)f3.x; a[13] = (_Float16)f3.y;
        a[14] = (_Float16)f3.z; a[15] = (_Float16)f3.w;
#pragma unroll
        for (int nt = 0; nt < 8; ++nt) {
            v16h b = wf[(ks * 8 + nt) * 32 + lane];
            acc[nt] = __builtin_amdgcn_wmma_f32_16x16x32_f16(
                false, a, false, b, (short)0, acc[nt], false, false);
        }
    }

#pragma unroll
    for (int nt = 0; nt < 8; ++nt) {
        int col = nt * 16 + lo;
        float bv = bias[col];
#pragma unroll
        for (int v = 0; v < 8; ++v) {
            int row = rowBase + hi * 8 + v;
            if (row < Mrows) {
                float val = acc[nt][v] + bv;
                if (apply_gelu) val = gelu_exact(val);
                out[(long)row * H_ + col] = val;
            }
        }
    }
}

// ---------------------------------------------------------------------------
// Pool scatter: per (c, n): g = batch[n], w = node_mask[c,n]
//   num[c,g,:] += O[c,n,:] * w ; den[c,g] += w
// ---------------------------------------------------------------------------
__global__ void pool_scatter_kernel(const float* __restrict__ O,
                                    const int*   __restrict__ batch,
                                    const float* __restrict__ nmask,
                                    float* __restrict__ num,
                                    float* __restrict__ den) {
    int wid  = threadIdx.x >> 5;
    int lane = threadIdx.x & 31;
    long id = (long)blockIdx.x * 8 + wid;
    if (id >= (long)C_ * N_) return;
    int c = (int)(id / N_);
    int n = (int)(id - (long)c * N_);
    int g = batch[n];
    float w = nmask[(long)c * N_ + n];
    const float4 o = *(const float4*)(O + ((long)(c * N_ + n)) * H_ + lane * 4);
    float* np = num + ((long)(c * G_ + g)) * F_ + lane * 4;
    atomic_add_f32(np + 0, o.x * w);
    atomic_add_f32(np + 1, o.y * w);
    atomic_add_f32(np + 2, o.z * w);
    atomic_add_f32(np + 3, o.w * w);
    if (lane == 0) atomic_add_f32(den + c * G_ + g, w);
}

// out[(g*C + c)*F + h] = num[c,g,h] / (den[c,g] + 1e-7)
__global__ void pool_final_kernel(const float* __restrict__ num,
                                  const float* __restrict__ den,
                                  float* __restrict__ out) {
    int t = blockIdx.x * 256 + threadIdx.x;   // 0..32767
    if (t >= G_ * C_ * F_) return;
    int h = t & 127;
    int c = (t >> 7) & 3;
    int g = t >> 9;
    out[t] = num[((long)(c * G_ + g)) * F_ + h] / (den[c * G_ + g] + 1e-7f);
}

// ---------------------------------------------------------------------------
extern "C" void kernel_launch(void* const* d_in, const int* in_sizes, int n_in,
                              void* d_out, int out_size, void* d_ws, size_t ws_size,
                              hipStream_t stream) {
    const float* x     = (const float*)d_in[0];
    const int*   batch = (const int*)d_in[1];
    const int*   ei    = (const int*)d_in[2];
    const float* ea    = (const float*)d_in[3];
    const float* nmask = (const float*)d_in[4];
    const float* emask = (const float*)d_in[5];
    const float* Wbe   = (const float*)d_in[6];
    const float* bbe   = (const float*)d_in[7];
    const float* eps   = (const float*)d_in[8];
    const float* W1    = (const float*)d_in[9];
    const float* b1    = (const float*)d_in[10];
    const float* W2    = (const float*)d_in[11];
    const float* b2    = (const float*)d_in[12];
    const float* Wm1   = (const float*)d_in[13];
    const float* bm1   = (const float*)d_in[14];
    const float* Wm2   = (const float*)d_in[15];
    const float* bm2   = (const float*)d_in[16];

    const size_t SZ   = (size_t)M_ * H_ * sizeof(float);   // 102,400,000 B
    const size_t WFN  = 4 * 8 * 32 * 16;                   // 16384 halfs/matrix
    char* ws = (char*)d_ws;
    float*    ee  = (float*)(ws);
    float*    B0  = (float*)(ws + SZ);
    float*    B1  = (float*)(ws + 2 * SZ);
    float*    B2  = (float*)(ws + 3 * SZ);
    _Float16* wf  = (_Float16*)(ws + 4 * SZ);
    float*    num = (float*)(ws + 4 * SZ + 6 * WFN * sizeof(_Float16));
    float*    den = num + C_ * G_ * F_;

    size_t need = 4 * SZ + 6 * WFN * sizeof(_Float16)
                + (size_t)(C_ * G_ * F_ + C_ * G_) * sizeof(float);
    if (ws_size < need) return;

    // 1) shuffle the six 128x128 weight matrices into WMMA fragment order
    prep_wfrag_kernel<<<64, 256, 0, stream>>>(W1,            wf + 0 * WFN);
    prep_wfrag_kernel<<<64, 256, 0, stream>>>(W1 + H_ * H_,  wf + 1 * WFN);
    prep_wfrag_kernel<<<64, 256, 0, stream>>>(W2,            wf + 2 * WFN);
    prep_wfrag_kernel<<<64, 256, 0, stream>>>(W2 + H_ * H_,  wf + 3 * WFN);
    prep_wfrag_kernel<<<64, 256, 0, stream>>>(Wm1,           wf + 4 * WFN);
    prep_wfrag_kernel<<<64, 256, 0, stream>>>(Wm2,           wf + 5 * WFN);

    // 2) bond encoder (shared across layers)
    bond_encoder_kernel<<<(E_ * H_) / 256, 256, 0, stream>>>(ea, Wbe, bbe, ee);

    // 3) conv layers
    const int gemmBlocks = (M_ + 127) / 128;   // 1563
    const float* xin = x;
    for (int l = 0; l < L_; ++l) {
        hipMemsetAsync(B0, 0, SZ, stream);
        scatter_msg_kernel<<<(C_ * E_) / 8, 256, 0, stream>>>(xin, ee, ei, emask, B0);
        residual_kernel<<<(M_ * H_ / 4) / 256, 256, 0, stream>>>(xin, eps, l, B0);
        gemm128_kernel<<<gemmBlocks, 256, 0, stream>>>(
            B0, wf + (size_t)l * WFN,       b1 + l * H_, B1, M_, 1);
        gemm128_kernel<<<gemmBlocks, 256, 0, stream>>>(
            B1, wf + (size_t)(2 + l) * WFN, b2 + l * H_, B2, M_, 1);
        xin = B2;   // safe: B2 only written after all reads of previous xin
    }

    // 4) final node MLP
    gemm128_kernel<<<gemmBlocks, 256, 0, stream>>>(B2, wf + 4 * WFN, bm1, B0, M_, 1);
    gemm128_kernel<<<gemmBlocks, 256, 0, stream>>>(B0, wf + 5 * WFN, bm2, B1, M_, 0);

    // 5) masked mean-pool per (c, graph) and transpose to (G*C, F)
    hipMemsetAsync(num, 0, (size_t)(C_ * G_ * F_ + C_ * G_) * sizeof(float), stream);
    pool_scatter_kernel<<<(C_ * N_) / 8, 256, 0, stream>>>(B1, batch, nmask, num, den);
    pool_final_kernel<<<(G_ * C_ * F_) / 256, 256, 0, stream>>>(num, den, (float*)d_out);
}
